// MultiModalMoEClassifier_59313498358197
// MI455X (gfx1250) — compile-verified, matrix operand order checked by reading, not run on hardware
//
#include <hip/hip_runtime.h>
#include <math.h>

// ---------------------------------------------------------------------------
// MultiModalMoEClassifier for MI455X (gfx1250, wave32, WMMA)
//
// Convs (~355 GFLOP hot path) are implicit GEMMs on v_wmma_f32_16x16x32_bf16:
// 128x64 output tile per 256-thread block, 8 waves each owning a 2x2 grid of
// 16x16 subtiles (4 WMMAs per K-chunk from 2 A + 2 B ds_load_b128 fragment
// gathers). Tiles are register-pipelined: chunk k+1's global loads are issued
// (branchless, clamped addresses) while chunk k's WMMAs run; packed bf16
// stores land as contiguous ds_store_b128. KSZ is a template parameter so the
// K->(ci,j) de-linearization is a constant mul-shift. f32->bf16 uses native
// compiler conversions so ISel can pick hardware cvt instructions.
// GroupNorm stats via wave shfl-reduce + f32 atomics; fused GN+GELU+residual
// epilogue; temporal GN + mean(T) commute into per-channel sums.
//
// Input pointer mapping assumes the harness flattens setup_inputs() leaves in
// dict-insertion order: emg, imu, demo, then params{emg{b1{cw,ng,nb,rw},b2,b3,
// tn_g,tn_b}, imu{...}, demo{...}, fus{...}, gate{...}, exp{...}} = 51 leaves.
// ---------------------------------------------------------------------------

typedef __attribute__((ext_vector_type(16))) __bf16 v16bf;
typedef __attribute__((ext_vector_type(8)))  float  v8f;
typedef __attribute__((ext_vector_type(2)))  __bf16 v2bf;

union FragBF { v16bf v; unsigned int u[8]; };
union AccF   { v8f   v; float        f[8]; };

// native conversions (RNE) -> lets ISel pick hw cvt (pk) instructions
__device__ inline unsigned int pack_bf2(float lo, float hi) {
  v2bf p;
  p[0] = (__bf16)lo;
  p[1] = (__bf16)hi;
  return __builtin_bit_cast(unsigned int, p);
}

__device__ inline float gelu_exact(float x) {
  return 0.5f * x * (1.0f + erff(x * 0.70710678118654752f));
}

// ---------------------------------------------------------------------------
// Implicit-GEMM conv1d:  Y[b,co,t] = sum_{ci,j} W[co,ci,j] * X[b,ci,t*s+j-pad]
// M=Cout (128/block), N=Tout (64/block), K=Cin*KSZ (chunks of 32).
// Wave w: M-bands 2*(w&3), 2*(w&3)+1 ; N-cols 2*(w>>2), 2*(w>>2)+1.
// Fixed staging slots: A row tid>>1 (16 k's), B col tid>>2 (8 k's).
// ---------------------------------------------------------------------------
template <int KSZ>
__global__ __launch_bounds__(256)
void conv1d_wmma(const float* __restrict__ X, const float* __restrict__ W,
                 float* __restrict__ Y, float* __restrict__ stats,
                 int Cin, int Cout, int stride, int pad, int Tin, int Tout)
{
  __shared__ __bf16 Alds[128 * 32];   // [m][k], k-pairs contiguous
  __shared__ __bf16 Blds[64 * 32];    // [n][k], n-major so frag gather = b128

  const int tid  = threadIdx.x;
  const int lane = tid & 31;
  const int wave = tid >> 5;
  const int t0   = blockIdx.x * 64;
  const int co0  = blockIdx.y * 128;
  const int bz   = blockIdx.z;
  const int Ktot = Cin * KSZ;

  const int mb0 = (wave & 3) * 32;      // first of two 16-row M-bands
  const int nc0 = (wave >> 2) * 32;     // first of two 16-col N-cols

  // fixed staging mapping (hoisted out of the K loop)
  const int mi_a = tid >> 1;            // A row 0..127
  const int ka0  = (tid & 1) * 16;      // A k-base (16 consecutive elems)
  const int n_b  = tid >> 2;            // B col 0..63
  const int kb0  = (tid & 3) * 8;       // B k-base (8 consecutive elems)
  const float* __restrict__ wrow = W + (size_t)(co0 + mi_a) * Ktot;
  const float* __restrict__ xrow = X + (size_t)bz * Cin * Tin;
  const int tb = (t0 + n_b) * stride - pad;

  float areg[16], breg[8];

  auto load_chunk = [&](int kc0) {
#pragma unroll
    for (int q = 0; q < 16; ++q) {
      int kg  = kc0 + ka0 + q;
      int kcl = (kg < Ktot) ? kg : (Ktot - 1);      // clamp: always-legal load
      float v = wrow[kcl];
      areg[q] = (kg < Ktot) ? v : 0.f;              // v_cndmask, no branches
    }
#pragma unroll
    for (int q = 0; q < 8; ++q) {
      int kg  = kc0 + kb0 + q;
      int kcl = (kg < Ktot) ? kg : (Ktot - 1);
      int ci  = kcl / KSZ;                          // constant divisor
      int j   = kcl - ci * KSZ;
      int tin = tb + j;
      bool ok = (kg < Ktot) && (tin >= 0) && (tin < Tin);
      int tcl = (tin < 0) ? 0 : ((tin >= Tin) ? (Tin - 1) : tin);
      float v = xrow[(size_t)ci * Tin + tcl];
      breg[q] = ok ? v : 0.f;
    }
  };

  AccF a00, a01, a10, a11;
#pragma unroll
  for (int i = 0; i < 8; ++i) { a00.f[i] = 0.f; a01.f[i] = 0.f; a10.f[i] = 0.f; a11.f[i] = 0.f; }

  unsigned int* Apk = reinterpret_cast<unsigned int*>(Alds);
  unsigned int* Bpk = reinterpret_cast<unsigned int*>(Blds);

  load_chunk(0);                                    // prologue

  for (int kc0 = 0; kc0 < Ktot; kc0 += 32) {
    // --- pack & store current chunk regs -> LDS (contiguous b128 stores) --
#pragma unroll
    for (int q = 0; q < 8; ++q)
      Apk[mi_a * 16 + (ka0 >> 1) + q] = pack_bf2(areg[2 * q], areg[2 * q + 1]);
#pragma unroll
    for (int q = 0; q < 4; ++q)
      Bpk[n_b * 16 + (kb0 >> 1) + q] = pack_bf2(breg[2 * q], breg[2 * q + 1]);
    __syncthreads();

    // --- issue next chunk's global loads; latency hidden behind WMMAs -----
    if (kc0 + 32 < Ktot) {
      __builtin_prefetch(&wrow[kc0 + 64], 0, 2);    // L2 prefetch 2 chunks out
      load_chunk(kc0 + 32);
    }

    // --- fragment gather per ISA 7.12.2 16-bit A/B K-pair striping --------
    const int half = lane >> 4;
    const int m0r  = (mb0 + (lane & 15)) << 5;      // row base in bf16 units
    const int m1r  = (mb0 + 16 + (lane & 15)) << 5;
    const int n0r  = (nc0 + (lane & 15)) << 5;
    const int n1r  = (nc0 + 16 + (lane & 15)) << 5;
    FragBF fa0, fa1, fb0, fb1;
#pragma unroll
    for (int i = 0; i < 8; ++i) {
      int kk = ((i < 4) ? 0 : 16) + half * 8 + (i & 3) * 2;
      fa0.u[i] = *reinterpret_cast<const unsigned int*>(&Alds[m0r + kk]);
      fa1.u[i] = *reinterpret_cast<const unsigned int*>(&Alds[m1r + kk]);
      fb0.u[i] = *reinterpret_cast<const unsigned int*>(&Blds[n0r + kk]);
      fb1.u[i] = *reinterpret_cast<const unsigned int*>(&Blds[n1r + kk]);
    }
    a00.v = __builtin_amdgcn_wmma_f32_16x16x32_bf16(false, fa0.v, false, fb0.v, (short)0, a00.v, false, false);
    a01.v = __builtin_amdgcn_wmma_f32_16x16x32_bf16(false, fa0.v, false, fb1.v, (short)0, a01.v, false, false);
    a10.v = __builtin_amdgcn_wmma_f32_16x16x32_bf16(false, fa1.v, false, fb0.v, (short)0, a10.v, false, false);
    a11.v = __builtin_amdgcn_wmma_f32_16x16x32_bf16(false, fa1.v, false, fb1.v, (short)0, a11.v, false, false);
    __syncthreads();
  }

  // --- store D: VGPR i -> M=i (lanes 0-15) / M=i+8 (lanes 16-31) -----------
  const int halfo = lane >> 4;
  const int c0    = t0 + nc0 + (lane & 15);
#pragma unroll
  for (int i = 0; i < 8; ++i) {
    int r0 = co0 + mb0 + i + halfo * 8;
    int r1 = r0 + 16;
    Y[((size_t)bz * Cout + r0) * Tout + c0]      = a00.f[i];
    Y[((size_t)bz * Cout + r0) * Tout + c0 + 16] = a01.f[i];
    Y[((size_t)bz * Cout + r1) * Tout + c0]      = a10.f[i];
    Y[((size_t)bz * Cout + r1) * Tout + c0 + 16] = a11.f[i];
  }

  // --- GroupNorm partials: wave's 32 rows are 32-aligned -> single group ---
  if (stats) {
    float s = 0.f, ss = 0.f;
#pragma unroll
    for (int i = 0; i < 8; ++i) {
      s  += a00.f[i] + a01.f[i] + a10.f[i] + a11.f[i];
      ss += a00.f[i]*a00.f[i] + a01.f[i]*a01.f[i] + a10.f[i]*a10.f[i] + a11.f[i]*a11.f[i];
    }
    for (int off = 16; off; off >>= 1) {
      s  += __shfl_xor(s,  off, 32);
      ss += __shfl_xor(ss, off, 32);
    }
    if (lane == 0) {
      int gs = Cout >> 3;                  // group size (32 or 64)
      int g  = (co0 + mb0) / gs;
      atomicAdd(&stats[((size_t)bz * 8 + g) * 2 + 0], s);
      atomicAdd(&stats[((size_t)bz * 8 + g) * 2 + 1], ss);
    }
  }
}

// ---------------------------------------------------------------------------
// Fused GroupNorm + GELU + residual-add epilogue (float4 vectorized).
// ---------------------------------------------------------------------------
__global__ __launch_bounds__(256)
void gn_gelu_add(const float* __restrict__ H, const float* __restrict__ R,
                 const float* __restrict__ stats,
                 const float* __restrict__ gam, const float* __restrict__ bet,
                 float* __restrict__ Y, int C, int T, int total4)
{
  int i4 = blockIdx.x * blockDim.x + threadIdx.x;
  if (i4 >= total4) return;
  size_t base = (size_t)i4 * 4;
  int c = (int)((base / T) % C);
  int b = (int)(base / ((size_t)C * T));
  int g = c / (C >> 3);
  float cnt = (float)((C >> 3) * T);
  float s  = stats[((size_t)b * 8 + g) * 2 + 0];
  float ss = stats[((size_t)b * 8 + g) * 2 + 1];
  float m   = s / cnt;
  float var = fmaxf(ss / cnt - m * m, 0.f);
  float inv = rsqrtf(var + 1e-5f);
  float ga = gam[c], be = bet[c];
  float4 h = *reinterpret_cast<const float4*>(H + base);
  float4 r = *reinterpret_cast<const float4*>(R + base);
  float4 o;
  o.x = gelu_exact((h.x - m) * inv * ga + be) + r.x;
  o.y = gelu_exact((h.y - m) * inv * ga + be) + r.y;
  o.z = gelu_exact((h.z - m) * inv * ga + be) + r.z;
  o.w = gelu_exact((h.w - m) * inv * ga + be) + r.w;
  *reinterpret_cast<float4*>(Y + base) = o;
}

// ---------------------------------------------------------------------------
// Per-(b,c) sum / sumsq over T (for temporal GN + mean-pool tail).
// ---------------------------------------------------------------------------
__global__ __launch_bounds__(256)
void chan_reduce(const float* __restrict__ Y, float* __restrict__ csum,
                 float* __restrict__ css, int T)
{
  __shared__ float rs[256], rss[256];
  int bc = blockIdx.x, tid = threadIdx.x;
  float s = 0.f, ss = 0.f;
  for (int t = tid; t < T; t += 256) {
    float v = Y[(size_t)bc * T + t];
    s += v; ss += v * v;
  }
  rs[tid] = s; rss[tid] = ss; __syncthreads();
  for (int off = 128; off; off >>= 1) {
    if (tid < off) { rs[tid] += rs[tid + off]; rss[tid] += rss[tid + off]; }
    __syncthreads();
  }
  if (tid == 0) { csum[bc] = rs[0]; css[bc] = rss[0]; }
}

// temporal GN is affine-constant over T, so mean(T) commutes:
// henc[b,c] = (csum[b,c]/T - m_g) * inv_g * tng[c] + tnb[c]
__global__ __launch_bounds__(256)
void finalize_enc(const float* __restrict__ csum, const float* __restrict__ css,
                  const float* __restrict__ tng, const float* __restrict__ tnb,
                  float* __restrict__ henc, int T)
{
  __shared__ float gsum[8], gss[8];
  int b = blockIdx.x, c = threadIdx.x;   // C == 256
  if (c < 8) { gsum[c] = 0.f; gss[c] = 0.f; }
  __syncthreads();
  float s = csum[b * 256 + c], ss = css[b * 256 + c];
  int g = c >> 5;
  atomicAdd(&gsum[g], s); atomicAdd(&gss[g], ss);
  __syncthreads();
  float cnt = 32.f * (float)T;
  float m   = gsum[g] / cnt;
  float var = fmaxf(gss[g] / cnt - m * m, 0.f);
  float inv = rsqrtf(var + 1e-5f);
  henc[b * 256 + c] = ((s / (float)T) - m) * inv * tng[c] + tnb[c];
}

// --------------------------- head: demo MLP --------------------------------
__global__ __launch_bounds__(32)
void demo_mlp(const float* __restrict__ demo,
              const float* __restrict__ lng, const float* __restrict__ lnb,
              const float* __restrict__ w1, const float* __restrict__ b1,
              const float* __restrict__ w2, const float* __restrict__ b2,
              float* __restrict__ dout)
{
  int b = blockIdx.x, l = threadIdx.x;
  float x = (l < 8) ? demo[b * 8 + l] : 0.f;
  float s = x, ss = x * x;
  for (int off = 16; off; off >>= 1) { s += __shfl_xor(s, off, 32); ss += __shfl_xor(ss, off, 32); }
  float m = s / 8.f, var = fmaxf(ss / 8.f - m * m, 0.f), inv = rsqrtf(var + 1e-5f);
  float xn = (l < 8) ? (x - m) * inv * lng[l] + lnb[l] : 0.f;
  float a = (l < 16) ? b1[l] : 0.f;
  for (int i = 0; i < 8; ++i) {
    float xi = __shfl(xn, i, 32);
    if (l < 16) a += xi * w1[l * 8 + i];
  }
  float d1 = gelu_exact(a);
  float a2 = (l < 16) ? b2[l] : 0.f;
  for (int i = 0; i < 16; ++i) {
    float di = __shfl(d1, i, 32);
    if (l < 16) a2 += di * w2[l * 16 + i];
  }
  if (l < 16) dout[b * 16 + l] = a2;
}

// --------------------------- head: fusion ----------------------------------
__global__ __launch_bounds__(256)
void fusion(const float* __restrict__ he, const float* __restrict__ hi,
            const float* __restrict__ dd,
            const float* __restrict__ ln1g, const float* __restrict__ ln1b,
            const float* __restrict__ Wf, const float* __restrict__ bf,
            const float* __restrict__ ln2g, const float* __restrict__ ln2b,
            float* __restrict__ hf)
{
  __shared__ float xn[528];
  __shared__ float red[256], red2[256];
  int b = blockIdx.x, tid = threadIdx.x;
  xn[tid]       = he[b * 256 + tid];
  xn[256 + tid] = hi[b * 256 + tid];
  if (tid < 16) xn[512 + tid] = dd[b * 16 + tid];
  __syncthreads();
  float s = 0.f, ss = 0.f;
  for (int i = tid; i < 528; i += 256) { float v = xn[i]; s += v; ss += v * v; }
  red[tid] = s; red2[tid] = ss; __syncthreads();
  for (int off = 128; off; off >>= 1) {
    if (tid < off) { red[tid] += red[tid + off]; red2[tid] += red2[tid + off]; }
    __syncthreads();
  }
  float m   = red[0] / 528.f;
  float var = fmaxf(red2[0] / 528.f - m * m, 0.f);
  float inv = rsqrtf(var + 1e-5f);
  __syncthreads();
  for (int i = tid; i < 528; i += 256) xn[i] = (xn[i] - m) * inv * ln1g[i] + ln1b[i];
  __syncthreads();
  float a = bf[tid];
  for (int i = 0; i < 528; ++i) a += xn[i] * Wf[(size_t)tid * 528 + i];
  a = gelu_exact(a);
  __syncthreads();
  red[tid] = a; red2[tid] = a * a; __syncthreads();
  for (int off = 128; off; off >>= 1) {
    if (tid < off) { red[tid] += red[tid + off]; red2[tid] += red2[tid + off]; }
    __syncthreads();
  }
  float m2 = red[0] / 256.f;
  float v2 = fmaxf(red2[0] / 256.f - m2 * m2, 0.f);
  float i2 = rsqrtf(v2 + 1e-5f);
  hf[b * 256 + tid] = (a - m2) * i2 * ln2g[tid] + ln2b[tid];
}

// --------------------------- head: gate (softmax + top-2) ------------------
__global__ __launch_bounds__(32)
void gate_topk(const float* __restrict__ hf, const float* __restrict__ Wg,
               const float* __restrict__ bg, float* __restrict__ wout)
{
  int b = blockIdx.x, l = threadIdx.x;
  float logit = -3.0e38f;
  if (l < 8) {
    float a = bg[l];
    for (int i = 0; i < 256; ++i) a += hf[b * 256 + i] * Wg[l * 256 + i];
    logit = a;
  }
  float mx = logit;
  for (int off = 16; off; off >>= 1) mx = fmaxf(mx, __shfl_xor(mx, off, 32));
  float e = (l < 8) ? __expf(logit - mx) : 0.f;
  float s = e;
  for (int off = 16; off; off >>= 1) s += __shfl_xor(s, off, 32);
  float p = e / s;
  float best1 = -1.f; int i1 = -1;
  for (int i = 0; i < 8; ++i) {
    float pi = __shfl(p, i, 32);
    if (pi > best1) { best1 = pi; i1 = i; }     // first occurrence on ties
  }
  float best2 = -1.f; int i2 = -1;
  for (int i = 0; i < 8; ++i) {
    float pi = __shfl(p, i, 32);
    if (i != i1 && pi > best2) { best2 = pi; i2 = i; }
  }
  float wm = (l == i1 || l == i2) ? p : 0.f;
  float sm = wm;
  for (int off = 16; off; off >>= 1) sm += __shfl_xor(sm, off, 32);
  if (l < 8) wout[b * 8 + l] = wm / fmaxf(sm, 1e-9f);
}

// --------------------------- head: experts ---------------------------------
__global__ __launch_bounds__(256)
void experts(const float* __restrict__ hf,
             const float* __restrict__ fc1w, const float* __restrict__ fc1b,
             const float* __restrict__ lng, const float* __restrict__ lnb,
             const float* __restrict__ fc2w, const float* __restrict__ fc2b,
             float* __restrict__ eout)
{
  __shared__ float red[256], red2[256], hsh[256];
  int be = blockIdx.x, b = be >> 3, e = be & 7, tid = threadIdx.x;
  hsh[tid] = hf[b * 256 + tid];
  __syncthreads();
  float z = fc1b[e * 256 + tid];
  const float* wrow = fc1w + ((size_t)e * 256 + tid) * 256;
  for (int i = 0; i < 256; ++i) z += hsh[i] * wrow[i];
  z = gelu_exact(z);
  red[tid] = z; red2[tid] = z * z; __syncthreads();
  for (int off = 128; off; off >>= 1) {
    if (tid < off) { red[tid] += red[tid + off]; red2[tid] += red2[tid + off]; }
    __syncthreads();
  }
  float m   = red[0] / 256.f;
  float var = fmaxf(red2[0] / 256.f - m * m, 0.f);
  float inv = rsqrtf(var + 1e-5f);
  z = (z - m) * inv * lng[e * 256 + tid] + lnb[e * 256 + tid];
  for (int c = 0; c < 10; ++c) {
    __syncthreads();
    red[tid] = z * fc2w[((size_t)e * 10 + c) * 256 + tid];
    __syncthreads();
    for (int off = 128; off; off >>= 1) {
      if (tid < off) red[tid] += red[tid + off];
      __syncthreads();
    }
    if (tid == 0) eout[(size_t)be * 10 + c] = red[0] + fc2b[e * 10 + c];
  }
}

__global__ __launch_bounds__(32)
void combine(const float* __restrict__ wg, const float* __restrict__ eout,
             float* __restrict__ out)
{
  int b = blockIdx.x, c = threadIdx.x;
  if (c < 10) {
    float s = 0.f;
    for (int e = 0; e < 8; ++e) s += wg[b * 8 + e] * eout[((size_t)b * 8 + e) * 10 + c];
    out[b * 10 + c] = s;
  }
}

// ---------------------------------------------------------------------------
static inline void launch_conv(const float* X, const float* W, float* Y,
                               float* stats, int Cin, int Cout, int ksz,
                               int stride, int pad, int Tin, int Tout,
                               hipStream_t stream)
{
  dim3 g(Tout / 64, Cout / 128, 128);
  switch (ksz) {
    case 7: conv1d_wmma<7><<<g, 256, 0, stream>>>(X, W, Y, stats, Cin, Cout, stride, pad, Tin, Tout); break;
    case 5: conv1d_wmma<5><<<g, 256, 0, stream>>>(X, W, Y, stats, Cin, Cout, stride, pad, Tin, Tout); break;
    case 3: conv1d_wmma<3><<<g, 256, 0, stream>>>(X, W, Y, stats, Cin, Cout, stride, pad, Tin, Tout); break;
    default: conv1d_wmma<1><<<g, 256, 0, stream>>>(X, W, Y, stats, Cin, Cout, stride, pad, Tin, Tout); break;
  }
}

extern "C" void kernel_launch(void* const* d_in, const int* in_sizes, int n_in,
                              void* d_out, int out_size, void* d_ws, size_t ws_size,
                              hipStream_t stream)
{
  (void)in_sizes; (void)n_in; (void)out_size; (void)ws_size;
  const int B = 128;

  const float* x_emg  = (const float*)d_in[0];
  const float* x_imu  = (const float*)d_in[1];
  const float* x_demo = (const float*)d_in[2];

  struct Blk { const float *cw, *ng, *nb, *rw; };
  struct Enc { Blk b1, b2, b3; const float *tng, *tnb; };
  int p = 3;
  auto getBlk = [&](void) { Blk k; k.cw = (const float*)d_in[p++]; k.ng = (const float*)d_in[p++];
                            k.nb = (const float*)d_in[p++]; k.rw = (const float*)d_in[p++]; return k; };
  auto getEnc = [&](void) { Enc E; E.b1 = getBlk(); E.b2 = getBlk(); E.b3 = getBlk();
                            E.tng = (const float*)d_in[p++]; E.tnb = (const float*)d_in[p++]; return E; };
  Enc enc_emg = getEnc();
  Enc enc_imu = getEnc();
  const float* dm_lng = (const float*)d_in[p++];
  const float* dm_lnb = (const float*)d_in[p++];
  const float* dm_w1  = (const float*)d_in[p++];
  const float* dm_b1  = (const float*)d_in[p++];
  const float* dm_w2  = (const float*)d_in[p++];
  const float* dm_b2  = (const float*)d_in[p++];
  const float* fu_l1g = (const float*)d_in[p++];
  const float* fu_l1b = (const float*)d_in[p++];
  const float* fu_w   = (const float*)d_in[p++];
  const float* fu_b   = (const float*)d_in[p++];
  const float* fu_l2g = (const float*)d_in[p++];
  const float* fu_l2b = (const float*)d_in[p++];
  const float* gt_w   = (const float*)d_in[p++];
  const float* gt_b   = (const float*)d_in[p++];
  const float* ex_f1w = (const float*)d_in[p++];
  const float* ex_f1b = (const float*)d_in[p++];
  const float* ex_lng = (const float*)d_in[p++];
  const float* ex_lnb = (const float*)d_in[p++];
  const float* ex_f2w = (const float*)d_in[p++];
  const float* ex_f2b = (const float*)d_in[p++];

  // workspace carve-out (max activation tensor = 128*256*1024 = 128*512*512 f32)
  size_t off = 0;
  auto alloc = [&](size_t nfloats) {
    float* ptr = (float*)((char*)d_ws + off);
    off += ((nfloats * sizeof(float) + 255) / 256) * 256;
    return ptr;
  };
  const size_t BIG = (size_t)B * 256 * 1024;   // 33.55M floats
  float* bufY  = alloc(BIG);
  float* bufH  = alloc(BIG);
  float* bufR  = alloc(BIG);
  float* stats = alloc((size_t)B * 8 * 2);
  float* csum  = alloc((size_t)B * 256);
  float* css   = alloc((size_t)B * 256);
  float* henc_e = alloc((size_t)B * 256);
  float* henc_i = alloc((size_t)B * 256);
  float* ddemo  = alloc((size_t)B * 16);
  float* hfus   = alloc((size_t)B * 256);
  float* wgate  = alloc((size_t)B * 8);
  float* eout   = alloc((size_t)B * 8 * 10);

  auto run_block = [&](const float* X, const Blk& blk, int Cin, int Cout,
                       int ksz, int stride, int Tin, float* Yout) {
    int pad  = ksz / 2;
    int Tout = (Tin + 2 * pad - ksz) / stride + 1;
    hipMemsetAsync(stats, 0, (size_t)B * 8 * 2 * sizeof(float), stream);
    launch_conv(X, blk.cw, bufH, stats, Cin, Cout, ksz, stride, pad, Tin, Tout, stream);
    launch_conv(X, blk.rw, bufR, nullptr, Cin, Cout, 1, stride, 0, Tin, Tout, stream);
    int total4 = (B * Cout * Tout) / 4;
    gn_gelu_add<<<(total4 + 255) / 256, 256, 0, stream>>>(
        bufH, bufR, stats, blk.ng, blk.nb, Yout, Cout, Tout, total4);
  };

  auto run_encoder = [&](const float* X, const Enc& E, int Cin0, float* henc) {
    run_block(X,    E.b1, Cin0, 256, 7, 2, 2048, bufY);  // -> (B,256,1024)
    run_block(bufY, E.b2, 256,  512, 5, 2, 1024, bufY);  // -> (B,512,512)
    run_block(bufY, E.b3, 512,  256, 3, 1, 512,  bufY);  // -> (B,256,512)
    chan_reduce<<<B * 256, 256, 0, stream>>>(bufY, csum, css, 512);
    finalize_enc<<<B, 256, 0, stream>>>(csum, css, E.tng, E.tnb, henc, 512);
  };

  run_encoder(x_emg, enc_emg, 16, henc_e);
  run_encoder(x_imu, enc_imu, 6,  henc_i);

  demo_mlp<<<B, 32, 0, stream>>>(x_demo, dm_lng, dm_lnb, dm_w1, dm_b1, dm_w2, dm_b2, ddemo);
  fusion<<<B, 256, 0, stream>>>(henc_e, henc_i, ddemo, fu_l1g, fu_l1b,
                                fu_w, fu_b, fu_l2g, fu_l2b, hfus);
  gate_topk<<<B, 32, 0, stream>>>(hfus, gt_w, gt_b, wgate);
  experts<<<B * 8, 256, 0, stream>>>(hfus, ex_f1w, ex_f1b, ex_lng, ex_lnb,
                                     ex_f2w, ex_f2b, eout);
  combine<<<B, 32, 0, stream>>>(wgate, eout, (float*)d_out);
}
